// BatchSoftmaxLoss_5978594476364
// MI455X (gfx1250) — compile-verified, matrix operand order checked by reading, not run on hardware
//
#include <hip/hip_runtime.h>
#include <math.h>

typedef __attribute__((ext_vector_type(4)))  float v4f;
typedef __attribute__((ext_vector_type(8)))  float v8f;
typedef __attribute__((ext_vector_type(8)))  __bf16 v8bf;
typedef __attribute__((ext_vector_type(16))) __bf16 v16bf;
typedef __attribute__((ext_vector_type(4)))  unsigned short us4;

#define B_N     4096
#define D_K     768
#define EPSV    1e-6f
#define ROWS_PB 32      // rows per block (2 WMMA row tiles)
#define LDAB    776     // LDS stride in bf16; 1552 B, % 256 == 16 -> bank-balanced b128 frag reads
#define STRIPS  4       // column strips (waves = 2 row-tiles x 4 strips)

// round-to-nearest-even fp32 -> bf16 (as ushort)
__device__ __forceinline__ unsigned short bf16_rne(float x) {
    unsigned int u = __float_as_uint(x);
    u += 0x7FFFu + ((u >> 16) & 1u);
    return (unsigned short)(u >> 16);
}
__device__ __forceinline__ float bf16_hi_f(unsigned short h) {
    return __uint_as_float(((unsigned int)h) << 16);
}

// ---------------- kernel 0: split P into bf16 hi/lo ----------------
__global__ __launch_bounds__(256)
void split_bf16_kernel(const float* __restrict__ P,
                       unsigned short* __restrict__ Phi,
                       unsigned short* __restrict__ Plo) {
    const int idx = blockIdx.x * 256 + threadIdx.x;     // float4 index
    v4f x = *(const v4f*)(P + (size_t)idx * 4);
    us4 h, l;
#pragma unroll
    for (int i = 0; i < 4; ++i) {
        unsigned short hh = bf16_rne(x[i]);
        h[i] = hh;
        l[i] = bf16_rne(x[i] - bf16_hi_f(hh));
    }
    *(us4*)(Phi + (size_t)idx * 4) = h;
    *(us4*)(Plo + (size_t)idx * 4) = l;
}

// ---------------- kernel 1: per-row statistics (fp32 exact) ----------------
// na[i] = ||a_i||^2 + 2*eps*sum(a_i) + D*eps^2 ; np[j] = ||p_j||^2 - 2*eps*sum(p_j)
__global__ __launch_bounds__(64)
void row_stats_kernel(const float* __restrict__ A, const float* __restrict__ P,
                      float* __restrict__ na, float* __restrict__ npv) {
    const int row   = blockIdx.x;
    const int which = blockIdx.y;
    const float* r = ((which == 0) ? A : P) + (size_t)row * D_K;
    const int t = threadIdx.x;
    float ss = 0.f, s = 0.f;
    for (int k = t; k < D_K; k += 64) {
        float x = r[k];
        ss += x * x; s += x;
    }
    __shared__ float red0[64];
    __shared__ float red1[64];
    red0[t] = ss; red1[t] = s;
    __syncthreads();
    for (int off = 32; off > 0; off >>= 1) {
        if (t < off) { red0[t] += red0[t + off]; red1[t] += red1[t + off]; }
        __syncthreads();
    }
    if (t == 0) {
        if (which == 0) na[row]  = red0[0] + 2.f * EPSV * red1[0] + (float)D_K * EPSV * EPSV;
        else            npv[row] = red0[0] - 2.f * EPSV * red1[0];
    }
}

// load one bf16 WMMA fragment: per lane two contiguous 16B chunks at K and K+16
__device__ __forceinline__ v16bf load_frag(const unsigned short* base, int k) {
    v8bf c0 = *(const v8bf*)(base + k);
    v8bf c1 = *(const v8bf*)(base + k + 16);
    return __builtin_shufflevector(c0, c1, 0,1,2,3,4,5,6,7,8,9,10,11,12,13,14,15);
}

// ---------------- kernel 2: fused bf16x3 WMMA GEMM + streaming row softmax ----------------
__global__ __launch_bounds__(256)
void fused_gemm_softmax_kernel(const float* __restrict__ A,
                               const unsigned short* __restrict__ Phi,
                               const unsigned short* __restrict__ Plo,
                               const float* __restrict__ na, const float* __restrict__ npv,
                               float* __restrict__ lossRow) {
    extern __shared__ char smem[];
    float* mPart = (float*)smem;                       // 32*64
    float* sPart = mPart + ROWS_PB * 64;               // 32*64
    float* lDiag = sPart + ROWS_PB * 64;               // 32 (pads to 16512 B)
    unsigned short* lAhi = (unsigned short*)(smem + 16512);
    unsigned short* lAlo = lAhi + ROWS_PB * LDAB;

    const int tid      = threadIdx.x;
    const int blockRow = blockIdx.x * ROWS_PB;

    // ---- stage A band: load fp32 once, split to bf16 hi/lo into padded LDS ----
    const int V4_PER_ROW = D_K / 4;                    // 192
    for (int idx = tid; idx < ROWS_PB * V4_PER_ROW; idx += 256) {
        int row = idx / V4_PER_ROW, c4 = idx % V4_PER_ROW;
        v4f x = *(const v4f*)(A + (size_t)(blockRow + row) * D_K + c4 * 4);
        us4 h, l;
#pragma unroll
        for (int i = 0; i < 4; ++i) {
            unsigned short hh = bf16_rne(x[i]);
            h[i] = hh;
            l[i] = bf16_rne(x[i] - bf16_hi_f(hh));
        }
        *(us4*)(lAhi + row * LDAB + c4 * 4) = h;
        *(us4*)(lAlo + row * LDAB + c4 * 4) = l;
    }
    __syncthreads();

    const int wave    = tid >> 5, lane = tid & 31;
    const int rowtile = wave >> 2;                     // 0..1
    const int strip   = wave & 3;                      // 0..3
    const int half    = lane >> 4;                     // half-wave -> K base offset 8
    const int l15     = lane & 15;

    const unsigned short* aHiBase = lAhi + (rowtile * 16 + l15) * LDAB + half * 8;
    const unsigned short* aLoBase = lAlo + (rowtile * 16 + l15) * LDAB + half * 8;

    float na8[8];
#pragma unroll
    for (int v = 0; v < 8; ++v)
        na8[v] = na[blockRow + rowtile * 16 + half * 8 + v];

    float m[8], s[8];
#pragma unroll
    for (int v = 0; v < 8; ++v) { m[v] = -INFINITY; s[v] = 0.f; }

    const int diagTile = blockIdx.x * (ROWS_PB / 16) + rowtile;
    const int NTILES   = B_N / 16;

    for (int j = strip; j < NTILES; j += STRIPS) {
        const int col = j * 16 + l15;                  // B mirrored: lane = N
        const unsigned short* bHiBase = Phi + (size_t)col * D_K + half * 8;
        const unsigned short* bLoBase = Plo + (size_t)col * D_K + half * 8;

        v8f acc0 = {0.f,0.f,0.f,0.f,0.f,0.f,0.f,0.f};  // hi*hi
        v8f acc1 = {0.f,0.f,0.f,0.f,0.f,0.f,0.f,0.f};  // cross terms

        // software pipeline: fragments for chunk k are loaded in iteration k, used in iteration k+32
        v16bf cBhi = load_frag(bHiBase, 0);
        v16bf cBlo = load_frag(bLoBase, 0);
        v16bf cAhi = load_frag(aHiBase, 0);
        v16bf cAlo = load_frag(aLoBase, 0);
#pragma unroll 4
        for (int k = 32; k < D_K; k += 32) {
            // issue next-chunk loads first (globals have the longest latency)
            v16bf nBhi = load_frag(bHiBase, k);
            v16bf nBlo = load_frag(bLoBase, k);
            v16bf nAhi = load_frag(aHiBase, k);
            v16bf nAlo = load_frag(aLoBase, k);
            acc0 = __builtin_amdgcn_wmma_f32_16x16x32_bf16(
                       false, cAhi, false, cBhi, (short)0, acc0, false, false);
            acc1 = __builtin_amdgcn_wmma_f32_16x16x32_bf16(
                       false, cAhi, false, cBlo, (short)0, acc1, false, false);
            acc1 = __builtin_amdgcn_wmma_f32_16x16x32_bf16(
                       false, cAlo, false, cBhi, (short)0, acc1, false, false);
            cAhi = nAhi; cAlo = nAlo; cBhi = nBhi; cBlo = nBlo;
        }
        // peeled final chunk
        acc0 = __builtin_amdgcn_wmma_f32_16x16x32_bf16(
                   false, cAhi, false, cBhi, (short)0, acc0, false, false);
        acc1 = __builtin_amdgcn_wmma_f32_16x16x32_bf16(
                   false, cAhi, false, cBlo, (short)0, acc1, false, false);
        acc1 = __builtin_amdgcn_wmma_f32_16x16x32_bf16(
                   false, cAlo, false, cBhi, (short)0, acc1, false, false);

        const float npc = npv[col];
#pragma unroll
        for (int v = 0; v < 8; ++v) {
            float g  = acc0[v] + acc1[v];
            float d2 = na8[v] + npc - 2.f * g;
            d2 = d2 > 0.f ? d2 : 0.f;
            float score = 50.f - __builtin_amdgcn_sqrtf(d2);   // bare v_sqrt_f32
            if (j == diagTile && l15 == half * 8 + v)
                lDiag[rowtile * 16 + half * 8 + v] = score;
            float mn = m[v] > score ? m[v] : score;            // online softmax
            s[v] = s[v] * __expf(m[v] - mn) + __expf(score - mn);
            m[v] = mn;
        }
    }

    // ---- publish per-lane partial (max, sumexp) states ----
#pragma unroll
    for (int v = 0; v < 8; ++v) {
        int rl   = rowtile * 16 + half * 8 + v;
        int slot = strip * 16 + l15;
        mPart[rl * 64 + slot] = m[v];
        sPart[rl * 64 + slot] = s[v];
    }
    __syncthreads();

    // ---- deterministic merge: one thread per row ----
    if (tid < ROWS_PB) {
        float M = -INFINITY, S = 0.f;
        for (int p = 0; p < 64; ++p) {
            float mm = mPart[tid * 64 + p], ss = sPart[tid * 64 + p];
            float mn = M > mm ? M : mm;
            S = S * __expf(M - mn) + ss * __expf(mm - mn);
            M = mn;
        }
        lossRow[blockRow + tid] = M + __logf(S) - lDiag[tid];
    }
}

// ---------------- kernel 3: deterministic mean ----------------
__global__ __launch_bounds__(256)
void mean_kernel(const float* __restrict__ lossRow, float* __restrict__ out) {
    __shared__ float red[256];
    int t = threadIdx.x;
    float s = 0.f;
    for (int i = t; i < B_N; i += 256) s += lossRow[i];
    red[t] = s;
    __syncthreads();
    for (int off = 128; off > 0; off >>= 1) {
        if (t < off) red[t] += red[t + off];
        __syncthreads();
    }
    if (t == 0) out[0] = red[0] * (1.f / (float)B_N);
}

extern "C" void kernel_launch(void* const* d_in, const int* in_sizes, int n_in,
                              void* d_out, int out_size, void* d_ws, size_t ws_size,
                              hipStream_t stream) {
    const float* A = (const float*)d_in[0];   // rep_anchor [4096, 768]
    const float* P = (const float*)d_in[1];   // rep_pos    [4096, 768]
    // d_in[2] (rep_neg) unused by the reference.

    float* ws      = (float*)d_ws;
    float* na      = ws;                                  // 4096
    float* npv     = ws + B_N;                            // 4096
    float* lossRow = ws + 2 * B_N;                        // 4096
    unsigned short* Phi = (unsigned short*)(ws + 3 * B_N);        // 4096*768
    unsigned short* Plo = Phi + (size_t)B_N * D_K;                // 4096*768
    float* out     = (float*)d_out;

    split_bf16_kernel<<<(B_N * D_K / 4) / 256, 256, 0, stream>>>(P, Phi, Plo);
    row_stats_kernel<<<dim3(B_N, 2), 64, 0, stream>>>(A, P, na, npv);

    const size_t smem = 16512 + 2 * (size_t)ROWS_PB * LDAB * sizeof(unsigned short);
    (void)hipFuncSetAttribute((const void*)fused_gemm_softmax_kernel,
                              hipFuncAttributeMaxDynamicSharedMemorySize, (int)smem);
    fused_gemm_softmax_kernel<<<B_N / ROWS_PB, 256, smem, stream>>>(
        A, Phi, Plo, na, npv, lossRow);

    mean_kernel<<<1, 256, 0, stream>>>(lossRow, out);
}
